// MiniTransformerLayer_50285477101766
// MI455X (gfx1250) — compile-verified, matrix operand order checked by reading, not run on hardware
//
#include <hip/hip_runtime.h>
#include <hip/hip_bf16.h>
#include <math.h>

#define HIDDEN   2048
#define HEADS    16
#define HEAD_DIM 128
#define FFN_DIM  4096
#define SEQ      2048
#define BATCH    2
#define TOKENS   (BATCH * SEQ)           // 4096
#define BHSD     ((size_t)BATCH * HEADS * SEQ * HEAD_DIM)  // 8388608
#define LN_EPS   1e-5f

typedef unsigned short u16;
typedef __attribute__((ext_vector_type(16))) __bf16 v16bf;
typedef __attribute__((ext_vector_type(8)))  float  v8f;
typedef __attribute__((ext_vector_type(4)))  int    v4i;

union FragU { uint4 u[2]; v16bf v; };

#ifndef __has_builtin
#define __has_builtin(x) 0
#endif

#if __has_builtin(__builtin_amdgcn_global_load_async_to_lds_b128)
#define HAVE_ASYNC_LDS 1
#define AS1 __attribute__((address_space(1)))
#define AS3 __attribute__((address_space(3)))
__device__ __forceinline__ void async_b128(const u16* g, u16* l) {
  __builtin_amdgcn_global_load_async_to_lds_b128((AS1 v4i*)g, (AS3 v4i*)l, 0, 0);
}
__device__ __forceinline__ void wait_async_all() {
  asm volatile("s_wait_asynccnt 0x0" ::: "memory");
}
#else
#define HAVE_ASYNC_LDS 0
__device__ __forceinline__ void wait_async_all() {}
#endif

__device__ __forceinline__ u16 f2bf(float x) {
  union { float f; unsigned u; } v; v.f = x;
  unsigned r = v.u + 0x7fffu + ((v.u >> 16) & 1u);   // round-to-nearest-even
  return (u16)(r >> 16);
}
__device__ __forceinline__ float bf2f(u16 h) {
  union { unsigned u; float f; } v; v.u = ((unsigned)h) << 16;
  return v.f;
}

// Load a 16x32 bf16 WMMA operand fragment (A-layout, or B as N-major rows of
// an N x K matrix). Per CDNA5 ISA: lane L holds row (L%16); lanes 0-15 hold
// K = [0..7, 16..23], lanes 16-31 hold K = [8..15, 24..31] -> two 16B loads.
__device__ __forceinline__ v16bf load_frag(const u16* base, int stride, int row, int k0, int lane) {
  const int half = (lane >> 4) & 1;
  const u16* p = base + (size_t)row * stride + k0 + half * 8;
  FragU fu;
  fu.u[0] = *(const uint4*)(p);
  fu.u[1] = *(const uint4*)(p + 16);
  return fu.v;
}

// ---------------------------------------------------------------- converts
__global__ void cvt_f32_bf16(const float* __restrict__ src, u16* __restrict__ dst, int n4) {
  int i = blockIdx.x * blockDim.x + threadIdx.x;
  if (i >= n4) return;
  float4 f = ((const float4*)src)[i];
  dst[i * 4 + 0] = f2bf(f.x);
  dst[i * 4 + 1] = f2bf(f.y);
  dst[i * 4 + 2] = f2bf(f.z);
  dst[i * 4 + 3] = f2bf(f.w);
}

// ---------------------------------------------------------------- layernorm
__global__ __launch_bounds__(256) void layernorm_k(const float* __restrict__ x,
                                                   const float* __restrict__ g,
                                                   const float* __restrict__ b,
                                                   u16* __restrict__ out) {
  const int row = blockIdx.x, t = threadIdx.x;
  const float* xr = x + (size_t)row * HIDDEN;
  float vals[8];
  float s1 = 0.f, s2 = 0.f;
#pragma unroll
  for (int i = 0; i < 8; ++i) {
    float v = xr[t + 256 * i];
    vals[i] = v; s1 += v; s2 += v * v;
  }
#pragma unroll
  for (int off = 16; off > 0; off >>= 1) {
    s1 += __shfl_down(s1, off, 32);
    s2 += __shfl_down(s2, off, 32);
  }
  __shared__ float red1[8], red2[8];
  const int wid = t >> 5, lane = t & 31;
  if (lane == 0) { red1[wid] = s1; red2[wid] = s2; }
  __syncthreads();
  if (t == 0) {
    float a = 0.f, c = 0.f;
#pragma unroll
    for (int i = 0; i < 8; ++i) { a += red1[i]; c += red2[i]; }
    red1[0] = a; red2[0] = c;
  }
  __syncthreads();
  const float mean = red1[0] * (1.0f / HIDDEN);
  const float var  = red2[0] * (1.0f / HIDDEN) - mean * mean;
  const float rstd = rsqrtf(var + LN_EPS);
  u16* orow = out + (size_t)row * HIDDEN;
#pragma unroll
  for (int i = 0; i < 8; ++i) {
    int c = t + 256 * i;
    orow[c] = f2bf((vals[i] - mean) * rstd * g[c] + b[c]);
  }
}

// ---------------------------------------------------------------- GEMM
// C[M,N] = A[M,K](bf16) x W[N,K](bf16)^T, fp32 accumulate via WMMA.
// Block tile 128(M) x 256(N), 8 waves in 2x4, each wave owns 64x64 (4x4 WMMA
// tiles => 16 wmma per K-step vs 8 fragment loads). K-step 32, double-buffered
// LDS staged via async global->LDS DMA (ASYNCcnt) when available.
#define EPI_F32   0
#define EPI_RESID 1
#define EPI_GELU  2
#define EPI_QKV   3

template <int EPI>
__global__ __launch_bounds__(256) void gemm_bf16(const u16* __restrict__ A,
                                                 const u16* __restrict__ W,
                                                 const float* __restrict__ resid,
                                                 float* __restrict__ outF,
                                                 u16* __restrict__ outB,
                                                 int M, int N, int K) {
  __shared__ u16 sA[2][128 * 40];   // 128 rows x 32 k, padded stride 40 (80B)
  __shared__ u16 sB[2][256 * 40];   // 256 rows x 32 k
  const int t = threadIdx.x;
  const int lane = t & 31, wid = t >> 5;
  const int nl = lane & 15;
  const int wave_m = wid >> 2, wave_n = wid & 3;      // 2 x 4 waves
  const int blockN = blockIdx.x * 256, blockM = blockIdx.y * 128;

  v8f acc[4][4] = {};

  const int arow = t >> 1, akc = (t & 1) * 16;        // A: 128 rows x 32k / 256 thr
  const u16* aSrc = A + (size_t)(blockM + arow) * K + akc;
  const u16* bSrc = W + (size_t)(blockN + t) * K;     // B: 1 row per thread

  auto stage = [&](int buf, int k0) {
    u16* sa = &sA[buf][arow * 40 + akc];
    u16* sb = &sB[buf][t * 40];
    const u16* ga = aSrc + k0;
    const u16* gb = bSrc + k0;
#if HAVE_ASYNC_LDS
    async_b128(ga, sa);
    async_b128(ga + 8, sa + 8);
#pragma unroll
    for (int i = 0; i < 4; ++i) async_b128(gb + i * 8, sb + i * 8);
#else
    *(uint4*)(sa)     = *(const uint4*)(ga);
    *(uint4*)(sa + 8) = *(const uint4*)(ga + 8);
#pragma unroll
    for (int i = 0; i < 4; ++i) *(uint4*)(sb + i * 8) = *(const uint4*)(gb + i * 8);
#endif
  };

  stage(0, 0);
  int buf = 0;
  for (int k0 = 0; k0 < K; k0 += 32, buf ^= 1) {
    wait_async_all();               // own async DMAs into 'buf' complete
    __syncthreads();                // all waves' staging of 'buf' visible
    if (k0 + 32 < K) stage(buf ^ 1, k0 + 32);   // prefetch next tile via DMA

    const u16* cA = sA[buf];
    const u16* cB = sB[buf];
    v16bf af[4], bfr[4];
#pragma unroll
    for (int i = 0; i < 4; ++i) {
      af[i]  = load_frag(cA, 40, wave_m * 64 + i * 16 + nl, 0, lane);
      bfr[i] = load_frag(cB, 40, wave_n * 64 + i * 16 + nl, 0, lane);
    }
#pragma unroll
    for (int ti = 0; ti < 4; ++ti)
#pragma unroll
      for (int tj = 0; tj < 4; ++tj)
        acc[ti][tj] = __builtin_amdgcn_wmma_f32_16x16x32_bf16(false, af[ti], false, bfr[tj],
                                                              (short)0, acc[ti][tj], false, false);
  }

#pragma unroll
  for (int ti = 0; ti < 4; ++ti)
#pragma unroll
    for (int tj = 0; tj < 4; ++tj)
#pragma unroll
      for (int r = 0; r < 8; ++r) {
        const int m = blockM + wave_m * 64 + ti * 16 + ((lane >> 4) << 3) + r;
        const int n = blockN + wave_n * 64 + tj * 16 + nl;
        const float v = acc[ti][tj][r];
        if constexpr (EPI == EPI_F32) {
          outF[(size_t)m * N + n] = v;
        } else if constexpr (EPI == EPI_RESID) {
          outF[(size_t)m * N + n] = v + resid[(size_t)m * N + n];
        } else if constexpr (EPI == EPI_GELU) {
          const float gel = 0.5f * v * (1.0f + erff(v * 0.70710678118654752f));
          outB[(size_t)m * N + n] = f2bf(gel);
        } else {  // EPI_QKV: scatter into [3][B][H][S][D] bf16
          const int which = n >> 11, rem = n & 2047;
          const int head = rem >> 7, d = rem & 127;
          const int bb = m >> 11, s = m & 2047;
          const size_t dst = ((((size_t)which * BATCH + bb) * HEADS + head) * SEQ + s) * HEAD_DIM + d;
          outB[dst] = f2bf(v);
        }
      }
}

// ---------------------------------------------------------------- RoPE (in place on bf16 Q,K)
__global__ void rope_k(u16* __restrict__ q, u16* __restrict__ k,
                       const float* __restrict__ pe, int npairs) {
  const int tid = blockIdx.x * blockDim.x + threadIdx.x;
  if (tid >= 2 * npairs) return;
  u16* T = (tid < npairs) ? q : k;
  const int p = (tid < npairs) ? tid : tid - npairs;
  const int i = p & 63;            // pair index within head_dim/2
  const int bhs = p >> 6;
  const int s = bhs & (SEQ - 1);
  const float c  = pe[s * HEAD_DIM + 2 * i];        // pe even cols (sin in source)
  const float sn = pe[s * HEAD_DIM + 2 * i + 1];    // pe odd cols
  const size_t base = (size_t)bhs * HEAD_DIM + 2 * i;
  const float a0 = bf2f(T[base]), a1 = bf2f(T[base + 1]);
  T[base]     = f2bf(a0 * c  - a1 * sn);
  T[base + 1] = f2bf(a0 * sn + a1 * c);
}

// ---------------------------------------------------------------- flash attention
// grid.x = (SEQ/128) * B*H. Block: 256 thr / 8 waves; each wave owns 16 full
// query rows so online-softmax row reductions stay inside one wave.
__global__ __launch_bounds__(256) void attn_k(const u16* __restrict__ Q,
                                              const u16* __restrict__ Kg,
                                              const u16* __restrict__ Vg,
                                              u16* __restrict__ out) {
  __shared__ u16 sK[64 * 136];    // 64 kv rows x 128 d (pad 8)
  __shared__ u16 sVt[128 * 72];   // transposed: 128 d x 64 kv (pad 8)
  __shared__ u16 sP[128 * 72];    // 128 q x 64 kv probs (pad 8)
  const int t = threadIdx.x, lane = t & 31, wid = t >> 5;
  const int nl = lane & 15, halfq = lane >> 4;
  const int qt = blockIdx.x & 15;          // SEQ/128 = 16 query tiles
  const int bh = blockIdx.x >> 4;          // 0..B*H-1
  const u16* Qb = Q  + (size_t)bh * SEQ * HEAD_DIM;
  const u16* Kb = Kg + (size_t)bh * SEQ * HEAD_DIM;
  const u16* Vb = Vg + (size_t)bh * SEQ * HEAD_DIM;

  // Q fragments held in registers for the whole pass (4 k-chunks of 32)
  v16bf qf[4];
  const int qrow = qt * 128 + wid * 16 + nl;
#pragma unroll
  for (int kc = 0; kc < 4; ++kc) qf[kc] = load_frag(Qb, HEAD_DIM, qrow, kc * 32, lane);

  v8f o[8] = {};
  float mrow[8], lrow[8];
#pragma unroll
  for (int r = 0; r < 8; ++r) { mrow[r] = -1e30f; lrow[r] = 0.0f; }
  const float scale = 0.088388347648318447f;   // 1/sqrt(128)

  for (int jt = 0; jt < SEQ / 64; ++jt) {
    __syncthreads();                           // prev iter done reading LDS
    {                                          // stage K tile: 64 x 128
      const int row = t >> 2, dc = (t & 3) * 32;
      const uint4* src = (const uint4*)(Kb + (size_t)(jt * 64 + row) * HEAD_DIM + dc);
      uint4* dst = (uint4*)&sK[row * 136 + dc];
#pragma unroll
      for (int i = 0; i < 4; ++i) dst[i] = src[i];
    }
#pragma unroll
    for (int i = 0; i < 4; ++i) {              // stage V transposed: d x kv
      const int l8 = t + 256 * i;
      const int sl = l8 >> 4, dchunk = (l8 & 15) * 8;
      union { uint4 q; u16 s[8]; } tmp;
      tmp.q = *(const uint4*)(Vb + (size_t)(jt * 64 + sl) * HEAD_DIM + dchunk);
#pragma unroll
      for (int e = 0; e < 8; ++e) sVt[(dchunk + e) * 72 + sl] = tmp.s[e];
    }
    __syncthreads();

    // scores: 16 rows x 64 cols per wave = 4 n-tiles, 4 k-steps
    v8f sc[4] = {};
#pragma unroll
    for (int kc = 0; kc < 4; ++kc)
#pragma unroll
      for (int j = 0; j < 4; ++j) {
        v16bf kfrag = load_frag(sK, 136, j * 16 + nl, kc * 32, lane);
        sc[j] = __builtin_amdgcn_wmma_f32_16x16x32_bf16(false, qf[kc], false, kfrag,
                                                        (short)0, sc[j], false, false);
      }
#pragma unroll
    for (int j = 0; j < 4; ++j) sc[j] *= scale;

    // online softmax (per row: reduce over 4 tiles + 16 lanes of the half)
    float alpha[8];
#pragma unroll
    for (int r = 0; r < 8; ++r) {
      float v = fmaxf(fmaxf(sc[0][r], sc[1][r]), fmaxf(sc[2][r], sc[3][r]));
#pragma unroll
      for (int off = 1; off < 16; off <<= 1) v = fmaxf(v, __shfl_xor(v, off, 32));
      const float mnew = fmaxf(mrow[r], v);
      alpha[r] = __expf(mrow[r] - mnew);
      mrow[r] = mnew;
      float ssum = 0.0f;
#pragma unroll
      for (int j = 0; j < 4; ++j) {
        const float p = __expf(sc[j][r] - mnew);
        sc[j][r] = p; ssum += p;
      }
#pragma unroll
      for (int off = 1; off < 16; off <<= 1) ssum += __shfl_xor(ssum, off, 32);
      lrow[r] = lrow[r] * alpha[r] + ssum;
    }
    v8f av;
#pragma unroll
    for (int r = 0; r < 8; ++r) av[r] = alpha[r];
#pragma unroll
    for (int jd = 0; jd < 8; ++jd) o[jd] *= av;   // rescale accumulated O

    // write P (C-layout -> LDS row-major) for A-operand reuse
#pragma unroll
    for (int j = 0; j < 4; ++j)
#pragma unroll
      for (int r = 0; r < 8; ++r)
        sP[(wid * 16 + halfq * 8 + r) * 72 + j * 16 + nl] = f2bf(sc[j][r]);
    __syncthreads();

    // O += P x V : 8 d-tiles, 2 k-steps over the 64 kv
#pragma unroll
    for (int kc = 0; kc < 2; ++kc) {
      v16bf pf = load_frag(sP, 72, wid * 16 + nl, kc * 32, lane);
#pragma unroll
      for (int jd = 0; jd < 8; ++jd) {
        v16bf vf = load_frag(sVt, 72, jd * 16 + nl, kc * 32, lane);
        o[jd] = __builtin_amdgcn_wmma_f32_16x16x32_bf16(false, pf, false, vf,
                                                        (short)0, o[jd], false, false);
      }
    }
  }

  // normalize and write [B,S,H*D] bf16 activation for the out-projection GEMM
  const int b = bh >> 4, h = bh & 15;
#pragma unroll
  for (int jd = 0; jd < 8; ++jd)
#pragma unroll
    for (int r = 0; r < 8; ++r) {
      const int s = qt * 128 + wid * 16 + halfq * 8 + r;
      const int col = h * HEAD_DIM + jd * 16 + nl;
      out[((size_t)b * SEQ + s) * HIDDEN + col] = f2bf(o[jd][r] / lrow[r]);
    }
}

// ---------------------------------------------------------------- launcher
extern "C" void kernel_launch(void* const* d_in, const int* in_sizes, int n_in,
                              void* d_out, int out_size, void* d_ws, size_t ws_size,
                              hipStream_t stream) {
  (void)in_sizes; (void)n_in; (void)out_size; (void)ws_size;
  const float* x     = (const float*)d_in[0];
  const float* pe    = (const float*)d_in[1];
  const float* w_qkv = (const float*)d_in[2];
  const float* w_out = (const float*)d_in[3];
  const float* w_fc1 = (const float*)d_in[4];
  const float* w_fc2 = (const float*)d_in[5];
  const float* g1    = (const float*)d_in[6];
  const float* b1    = (const float*)d_in[7];
  const float* g2    = (const float*)d_in[8];
  const float* b2    = (const float*)d_in[9];
  float* out = (float*)d_out;

  char* ws = (char*)d_ws;
  size_t off = 0;
  auto alloc = [&](size_t bytes) -> void* {
    void* p = ws + off;
    off += (bytes + 255) & ~(size_t)255;
    return p;
  };

  u16*   wqkv_b = (u16*)alloc((size_t)3 * HIDDEN * HIDDEN * 2);
  u16*   wout_b = (u16*)alloc((size_t)HIDDEN * HIDDEN * 2);
  u16*   wfc1_b = (u16*)alloc((size_t)FFN_DIM * HIDDEN * 2);
  u16*   wfc2_b = (u16*)alloc((size_t)HIDDEN * FFN_DIM * 2);
  u16*   xn     = (u16*)alloc((size_t)TOKENS * HIDDEN * 2);    // LN output (reused)
  u16*   qkv    = (u16*)alloc(3 * BHSD * 2);                   // [3][B][H][S][D]
  u16*   attn_a = (u16*)alloc((size_t)TOKENS * HIDDEN * 2);
  float* x1     = (float*)alloc((size_t)TOKENS * HIDDEN * 4);  // post-attn residual
  u16*   hact   = (u16*)alloc((size_t)TOKENS * FFN_DIM * 2);

  // 1) weight conversions fp32 -> bf16
  {
    int n4;
    n4 = 3 * HIDDEN * HIDDEN / 4; cvt_f32_bf16<<<n4 / 256, 256, 0, stream>>>(w_qkv, wqkv_b, n4);
    n4 = HIDDEN * HIDDEN / 4;     cvt_f32_bf16<<<n4 / 256, 256, 0, stream>>>(w_out, wout_b, n4);
    n4 = FFN_DIM * HIDDEN / 4;    cvt_f32_bf16<<<n4 / 256, 256, 0, stream>>>(w_fc1, wfc1_b, n4);
    n4 = HIDDEN * FFN_DIM / 4;    cvt_f32_bf16<<<n4 / 256, 256, 0, stream>>>(w_fc2, wfc2_b, n4);
  }
  // 2) LN1
  layernorm_k<<<TOKENS, 256, 0, stream>>>(x, g1, b1, xn);
  // 3) QKV projection with scatter into [3,B,H,S,D]
  gemm_bf16<EPI_QKV><<<dim3((3 * HIDDEN) / 256, TOKENS / 128), 256, 0, stream>>>(
      xn, wqkv_b, nullptr, nullptr, qkv, TOKENS, 3 * HIDDEN, HIDDEN);
  // 4) RoPE in place on Q and K
  {
    const int npairs = BATCH * HEADS * SEQ * (HEAD_DIM / 2);   // 4194304
    rope_k<<<(2 * npairs) / 256, 256, 0, stream>>>(qkv, qkv + BHSD, pe, npairs);
  }
  // 5) attention (flash-style, full softmax over SEQ)
  attn_k<<<(SEQ / 128) * BATCH * HEADS, 256, 0, stream>>>(
      qkv, qkv + BHSD, qkv + 2 * BHSD, attn_a);
  // 6) output projection + residual -> x1
  gemm_bf16<EPI_RESID><<<dim3(HIDDEN / 256, TOKENS / 128), 256, 0, stream>>>(
      attn_a, wout_b, x, x1, nullptr, TOKENS, HIDDEN, HIDDEN);
  // 7) LN2
  layernorm_k<<<TOKENS, 256, 0, stream>>>(x1, g2, b2, xn);
  // 8) fc1 + exact GELU -> bf16
  gemm_bf16<EPI_GELU><<<dim3(FFN_DIM / 256, TOKENS / 128), 256, 0, stream>>>(
      xn, wfc1_b, nullptr, nullptr, hact, TOKENS, FFN_DIM, HIDDEN);
  // 9) fc2 + residual -> d_out
  gemm_bf16<EPI_RESID><<<dim3(HIDDEN / 256, TOKENS / 128), 256, 0, stream>>>(
      hact, wfc2_b, x1, out, nullptr, TOKENS, HIDDEN, FFN_DIM);
}
